// GATv2_40673340293773
// MI455X (gfx1250) — matrix operand order repaired and checked by reading, not compiled, not run on hardware
//
#include <hip/hip_runtime.h>
#include <stdint.h>

// ---------- CDNA5 WMMA types ----------
typedef __attribute__((ext_vector_type(16))) __bf16        v16bf;
typedef __attribute__((ext_vector_type(8)))  float         v8f;
typedef __attribute__((ext_vector_type(4)))  unsigned int  v4u;
typedef __attribute__((ext_vector_type(8)))  unsigned int  v8u;

// ---------- helpers ----------
__device__ __forceinline__ uint16_t f2bf(float f) {        // fp32 -> bf16 RNE
  unsigned u = __float_as_uint(f);
  unsigned r = 0x7FFFu + ((u >> 16) & 1u);
  return (uint16_t)((u + r) >> 16);
}
// monotonic f32 <-> u32 encoding so atomicMax(uint) == float max
__device__ __forceinline__ unsigned fenc(float f) {
  unsigned u = __float_as_uint(f);
  return (u & 0x80000000u) ? ~u : (u | 0x80000000u);
}
__device__ __forceinline__ float fdec(unsigned u) {
  unsigned b = (u & 0x80000000u) ? (u & 0x7FFFFFFFu) : ~u;
  return __uint_as_float(b);
}
__device__ __forceinline__ float mish_f(float x) {
  float sp = (x > 15.f) ? x : log1pf(__expf(x));           // softplus
  return x * tanhf(sp);
}

// ---------- 1) fuse + convert weights: WT[192][64] bf16 (row = output col), bcat[192] ----------
__global__ void gat_prep_weights(const float* __restrict__ We,  const float* __restrict__ Web,
                                 const float* __restrict__ Ws,  const float* __restrict__ Wsb,
                                 const float* __restrict__ Wr,  const float* __restrict__ Wrb,
                                 uint16_t* __restrict__ WT, float* __restrict__ bcat) {
  for (int idx = threadIdx.x; idx < 192 * 64; idx += blockDim.x) {
    int col = idx >> 6, k = idx & 63;
    float w;
    if (col < 64) {                      // P section: We[k][col]
      w = We[k * 64 + col];
    } else if (col < 128) {              // S section: (We @ Ws)[k][col-64]
      int n = col - 64; float acc = 0.f;
      for (int j = 0; j < 64; ++j) acc += We[k * 64 + j] * Ws[j * 64 + n];
      w = acc;
    } else {                             // R section: Wr[k][col-128]
      w = Wr[k * 64 + (col - 128)];
    }
    WT[idx] = f2bf(w);
  }
  for (int n = threadIdx.x; n < 192; n += blockDim.x) {
    float b;
    if (n < 64) b = Web[n];
    else if (n < 128) {
      int c = n - 64; float acc = Wsb[c];
      for (int j = 0; j < 64; ++j) acc += Web[j] * Ws[j * 64 + c];
      b = acc;
    } else b = Wrb[n - 128];
    bcat[n] = b;
  }
}

// ---------- 2) nodes fp32 -> bf16 ----------
__global__ void gat_convert_nodes(const float* __restrict__ nodes, uint16_t* __restrict__ Xb, int tot) {
  for (int i = blockIdx.x * blockDim.x + threadIdx.x; i < tot; i += gridDim.x * blockDim.x)
    Xb[i] = f2bf(nodes[i]);
}

// ---------- 3) zero out / init segmax, denom ----------
__global__ void gat_init(float* __restrict__ out, unsigned* __restrict__ segmax,
                         float* __restrict__ denom, int N) {
  int tot = N * 64;
  for (int i = blockIdx.x * blockDim.x + threadIdx.x; i < tot; i += gridDim.x * blockDim.x) {
    out[i] = 0.f;
    if (i < N) { segmax[i] = 0u; denom[i] = 0.f; }   // fenc-space 0 == "-infinity"
  }
}

// ---------- 4) WMMA GEMM: [N x 64] @ [64 x 192] -> P|S|R, fp32 accumulate ----------
// One wave computes a 16-row tile across all 192 output columns (12 n-tiles x 2 k-steps = 24 WMMAs).
__global__ __launch_bounds__(256) void gat_gemm_psr(
    const uint16_t* __restrict__ Xb, const uint16_t* __restrict__ WT,
    const float* __restrict__ bcat,
    float* __restrict__ P, float* __restrict__ S, float* __restrict__ R, int N) {
  int lane  = threadIdx.x & 31;
  int tile  = blockIdx.x * (blockDim.x >> 5) + (threadIdx.x >> 5);
  int tiles = (N + 15) >> 4;
  if (tile >= tiles) return;
  int mcol = lane & 15;          // A row / B col / C col index per ISA layout
  int half = lane >> 4;
  bool full = (tile * 16 + 16) <= N;   // only the last tile can be partial

  int row  = tile * 16 + mcol;
  int rowc = row < N ? row : N - 1;
  const uint16_t* xr = Xb + (size_t)rowc * 64;
  // A fragment (16x32 bf16): K = 16*half + e
  v4u a0lo = *(const v4u*)(xr + 16 * half);
  v4u a0hi = *(const v4u*)(xr + 8 + 16 * half);
  v4u a1lo = *(const v4u*)(xr + 32 + 16 * half);
  v4u a1hi = *(const v4u*)(xr + 40 + 16 * half);
  v8u a0r = __builtin_shufflevector(a0lo, a0hi, 0, 1, 2, 3, 4, 5, 6, 7);
  v8u a1r = __builtin_shufflevector(a1lo, a1hi, 0, 1, 2, 3, 4, 5, 6, 7);
  v16bf a0 = __builtin_bit_cast(v16bf, a0r);
  v16bf a1 = __builtin_bit_cast(v16bf, a1r);

  float* outs[3] = { P, S, R };
  int rbase = tile * 16 + 8 * half;    // first C row this lane owns
  #pragma unroll
  for (int t = 0; t < 12; ++t) {
    int gcol = t * 16 + mcol;                       // 0..191 global output column
    const uint16_t* wrow = WT + (size_t)gcol * 64;  // 64 k-values for this column
    // B fragment (32x16 bf16): lane col = lane&15, K = 16*(lane>=16) + e
    v16bf b0 = __builtin_bit_cast(v16bf, *(const v8u*)(wrow + 16 * half));
    v16bf b1 = __builtin_bit_cast(v16bf, *(const v8u*)(wrow + 32 + 16 * half));
    v8f c = {};
    c = __builtin_amdgcn_wmma_f32_16x16x32_bf16(false, a0, false, b0, (short)0, c, false, false);
    c = __builtin_amdgcn_wmma_f32_16x16x32_bf16(false, a1, false, b1, (short)0, c, false, false);
    float bias = bcat[gcol];
    int ncol = (t & 3) * 16 + mcol;                 // column within 64-wide output
    // C layout: element v -> row rbase+v, col = lane&15
    float* dstp = outs[t >> 2] + (size_t)rbase * 64 + ncol;
    if (full) {                                     // unguarded: immediate-offset stores
      #pragma unroll
      for (int v = 0; v < 8; ++v) dstp[(size_t)v * 64] = c[v] + bias;
    } else {                                        // rare partial tail tile
      #pragma unroll
      for (int v = 0; v < 8; ++v)
        if (rbase + v < N) dstp[(size_t)v * 64] = c[v] + bias;
    }
  }
}

// ---------- 5) per-edge logits + segment max (one wave per edge) ----------
__global__ __launch_bounds__(256) void gat_edge_logits(
    const int* __restrict__ senders, const int* __restrict__ receivers,
    const float* __restrict__ S, const float* __restrict__ R,
    const float* __restrict__ Wa, const float* __restrict__ Wab,
    float* __restrict__ logits, unsigned* __restrict__ segmax, int E) {
  int lane   = threadIdx.x & 31;
  int wave   = (blockIdx.x * blockDim.x + threadIdx.x) >> 5;
  int nwaves = (gridDim.x * blockDim.x) >> 5;
  float wa0 = Wa[lane], wa1 = Wa[lane + 32], wab = Wab[0];
  for (int e = wave; e < E; e += nwaves) {
    int s = senders[e], r = receivers[e];
    const float* sr = S + (size_t)s * 64;
    const float* rr = R + (size_t)r * 64;
    float x0 = sr[lane]      + rr[lane];
    float x1 = sr[lane + 32] + rr[lane + 32];
    float p  = mish_f(x0) * wa0 + mish_f(x1) * wa1;
    #pragma unroll
    for (int off = 16; off; off >>= 1) p += __shfl_xor(p, off, 32);
    if (lane == 0) {
      float lg = p + wab;
      logits[e] = lg;
      atomicMax(segmax + r, fenc(lg));
    }
  }
}

// ---------- 6) per-edge exp + accumulate (denom & un-normalized output) ----------
__global__ __launch_bounds__(256) void gat_edge_accum(
    const int* __restrict__ senders, const int* __restrict__ receivers,
    const float* __restrict__ P, const float* __restrict__ logits,
    const unsigned* __restrict__ segmax, float* __restrict__ denom,
    float* __restrict__ out, int E) {
  int lane   = threadIdx.x & 31;
  int wave   = (blockIdx.x * blockDim.x + threadIdx.x) >> 5;
  int nwaves = (gridDim.x * blockDim.x) >> 5;
  for (int e = wave; e < E; e += nwaves) {
    int s = senders[e], r = receivers[e];
    float u = __expf(logits[e] - fdec(segmax[r]));   // broadcast loads, same u in all lanes
    if (lane == 0) unsafeAtomicAdd(denom + r, u);
    const float* pr = P + (size_t)s * 64;
    float* orow = out + (size_t)r * 64;
    unsafeAtomicAdd(orow + lane,      u * pr[lane]);
    unsafeAtomicAdd(orow + lane + 32, u * pr[lane + 32]);
  }
}

// ---------- 7) divide by softmax denominator ----------
__global__ void gat_finalize(float* __restrict__ out, const float* __restrict__ denom, int N) {
  int tot = N * 64;
  for (int i = blockIdx.x * blockDim.x + threadIdx.x; i < tot; i += gridDim.x * blockDim.x) {
    float d = denom[i >> 6];
    out[i] = (d > 0.f) ? out[i] / d : 0.f;
  }
}

extern "C" void kernel_launch(void* const* d_in, const int* in_sizes, int n_in,
                              void* d_out, int out_size, void* d_ws, size_t ws_size,
                              hipStream_t stream) {
  const float* nodes     = (const float*)d_in[0];
  const int*   senders   = (const int*)  d_in[1];
  const int*   receivers = (const int*)  d_in[2];
  const float* We  = (const float*)d_in[3];
  const float* Web = (const float*)d_in[4];
  const float* Ws  = (const float*)d_in[5];
  const float* Wsb = (const float*)d_in[6];
  const float* Wr  = (const float*)d_in[7];
  const float* Wrb = (const float*)d_in[8];
  const float* Wa  = (const float*)d_in[9];
  const float* Wab = (const float*)d_in[10];
  float* out = (float*)d_out;
  int N = in_sizes[0] / 64;
  int E = in_sizes[1];
  (void)n_in; (void)out_size; (void)ws_size;

  char* ws = (char*)d_ws;
  size_t off = 0;
  auto take = [&](size_t b) -> char* {
    char* p = ws + off; off += (b + 255) & ~(size_t)255; return p;
  };
  uint16_t* Xb     = (uint16_t*)take((size_t)N * 64 * 2);   // bf16 nodes
  uint16_t* WT     = (uint16_t*)take(192 * 64 * 2);         // fused bf16 weights, col-major
  float*    bcat   = (float*)   take(192 * 4);
  float*    P      = (float*)   take((size_t)N * 64 * 4);
  float*    S      = (float*)   take((size_t)N * 64 * 4);
  float*    R      = (float*)   take((size_t)N * 64 * 4);
  float*    logits = (float*)   take((size_t)E * 4);
  unsigned* segmax = (unsigned*)take((size_t)N * 4);
  float*    denom  = (float*)   take((size_t)N * 4);

  gat_prep_weights<<<1, 256, 0, stream>>>(We, Web, Ws, Wsb, Wr, Wrb, WT, bcat);
  gat_convert_nodes<<<1024, 256, 0, stream>>>(nodes, Xb, N * 64);
  gat_init<<<1024, 256, 0, stream>>>(out, segmax, denom, N);

  int tiles = (N + 15) >> 4;
  gat_gemm_psr<<<(tiles + 7) / 8, 256, 0, stream>>>(Xb, WT, bcat, P, S, R, N);

  int eblocks = (E + 7) / 8;   // one wave per edge, 8 waves per 256-thread block
  gat_edge_logits<<<eblocks, 256, 0, stream>>>(senders, receivers, S, R, Wa, Wab, logits, segmax, E);
  gat_edge_accum<<<eblocks, 256, 0, stream>>>(senders, receivers, P, logits, segmax, denom, out, E);
  gat_finalize<<<1024, 256, 0, stream>>>(out, denom, N);
}
